// NSFPRawMLP_4355096838685
// MI455X (gfx1250) — matrix operand order repaired and checked by reading, not compiled, not run on hardware
//
#include <hip/hip_runtime.h>
#include <hip/hip_bf16.h>

typedef _Float16 v16h __attribute__((ext_vector_type(16)));
typedef _Float16 v8h  __attribute__((ext_vector_type(8)));
typedef float    v8f  __attribute__((ext_vector_type(8)));

#define N_HIDDEN 7
#define LATENT   128
#define HSTR     136   // 128 + 8-half pad -> row stride 272B (odd multiple of 16B): conflict-free b128
#define BLK      256   // 8 waves, 256 points per block

// LDS plan (halfs): [ Wt: 7*128*HSTR = 121856 ][ H: 256*HSTR = 34816 ]  => 313,344 bytes <= 320KB
#define WT_HALFS (N_HIDDEN * LATENT * HSTR)
#define LDS_HALFS (WT_HALFS + BLK * HSTR)

// ---------- prep: w_hid [l][k][n] (f32) -> wt [l][n][k] (f16, compact 128-stride) == W^T ----------
__global__ void xpose_weights_f16(const float* __restrict__ w_hid, _Float16* __restrict__ wt) {
    int i = blockIdx.x * 256 + threadIdx.x;            // 0 .. 7*128*128-1
    if (i >= N_HIDDEN * LATENT * LATENT) return;
    int l = i >> 14;
    int r = i & 16383;
    int n = r >> 7;
    int k = r & 127;
    wt[i] = (_Float16)w_hid[(l << 14) + (k << 7) + n]; // write coalesced over k
}

// ---------- main fused MLP: per layer compute O = W^T x h^T so both activation
// reads (B operand) and activation writes (D rows) are contiguous b128 in LDS ----------
__global__ __launch_bounds__(BLK) void mlp_wmma_kernel(
    const float* __restrict__ x,     const float* __restrict__ w_in,
    const float* __restrict__ b_in,  const float* __restrict__ b_hid,
    const float* __restrict__ w_out, const float* __restrict__ b_out,
    const _Float16* __restrict__ wt_g, float* __restrict__ out)
{
    extern __shared__ _Float16 smem[];
    _Float16* Wt = smem;            // padded: row n of layer l at Wt + (l*128+n)*HSTR (== W^T rows)
    _Float16* H  = smem + WT_HALFS; // padded: point p's features at H + p*HSTR

    const int t = threadIdx.x;
    const v8h hzero = {};

    // --- cooperative weight preload: compact global f16 -> padded LDS, 16B chunks ---
    for (int i = t; i < N_HIDDEN * LATENT * (LATENT / 8); i += BLK) { // 14336 chunks of 8 halfs
        int row = i >> 4;          // l*128 + n
        int c8  = i & 15;          // which 8-half chunk of the 128-half row
        uint4 v = ((const uint4*)wt_g)[i];
        *(uint4*)(Wt + row * HSTR + c8 * 8) = v;
    }

    // --- input layer: Linear(3,128)+ReLU, thread t owns point/row t; packed b128 stores ---
    {
        const int p = blockIdx.x * BLK + t;
        const float x0 = x[p * 3 + 0], x1 = x[p * 3 + 1], x2 = x[p * 3 + 2];
        _Float16* hr = H + t * HSTR;
        for (int j = 0; j < LATENT; j += 8) {
            union { v8h v; _Float16 e[8]; } o;
            #pragma unroll
            for (int e = 0; e < 8; ++e) {
                float v = fmaf(x0, w_in[j + e], fmaf(x1, w_in[LATENT + j + e],
                          fmaf(x2, w_in[2 * LATENT + j + e], b_in[j + e])));
                o.e[e] = (_Float16)v;            // cvt first (canonical result) ...
            }
            o.v = __builtin_elementwise_max(o.v, hzero);  // ... then packed f16 ReLU
            *(v8h*)(hr + j) = o.v;
        }
    }
    // Only cross-wave dependency in the whole kernel: Wt preload -> Wt reads.
    __syncthreads();

    const int lane = t & 31;
    const int wv   = t >> 5;       // wave id: owns points [32*wv, 32*wv+32)
    const int half = lane >> 4;    // 0: lanes 0-15, 1: lanes 16-31
    const int l16  = lane & 15;

    // --- 7 hidden layers (no barriers: H-row deps are wave-local, DS is in-order per wave) ---
    for (int l = 0; l < N_HIDDEN; ++l) {
        const _Float16* Wl = Wt + l * LATENT * HSTR;

        // Preload ALL activation fragments (B = h^T) for this wave's 32 points.
        // B layout: lane holds K = kc*32 + {8*half + 0..7, 16 + 8*half + 0..7} for point l16.
        v16h Bfrag[2][4];
        #pragma unroll
        for (int nt = 0; nt < 2; ++nt) {
            const int p = wv * 32 + nt * 16 + l16;
            #pragma unroll
            for (int kc = 0; kc < 4; ++kc) {
                const _Float16* pb = H + p * HSTR + kc * 32 + half * 8;
                union { v16h v; v8h q[2]; } u;
                u.q[0] = *(const v8h*)(pb);
                u.q[1] = *(const v8h*)(pb + 16);
                Bfrag[nt][kc] = u.v;
            }
        }

        #pragma unroll
        for (int mt = 0; mt < 8; ++mt) {
            // A = W^T tile: lane row M = l16, K blocks {0..7,16..23} / {8..15,24..31}
            v16h Afrag[4];
            #pragma unroll
            for (int kc = 0; kc < 4; ++kc) {
                const _Float16* pa = Wl + (mt * 16 + l16) * HSTR + kc * 32 + half * 8;
                union { v16h v; v8h q[2]; } u;
                u.q[0] = *(const v8h*)(pa);
                u.q[1] = *(const v8h*)(pa + 16);
                Afrag[kc] = u.v;
            }
            // bias folded into accumulator init: VGPR r holds feature mt*16 + 8*half + r
            const v8f bias = *(const v8f*)(b_hid + l * LATENT + mt * 16 + 8 * half);

            #pragma unroll
            for (int nt = 0; nt < 2; ++nt) {
                v8f acc = bias;
                #pragma unroll
                for (int kc = 0; kc < 4; ++kc) {
                    acc = __builtin_amdgcn_wmma_f32_16x16x32_f16(
                        false, Afrag[kc], false, Bfrag[nt][kc], (short)0, acc, false, false);
                }
                // D: lane = point (N=l16), VGPRs = 8 consecutive features -> one packed b128 store
                const int p = wv * 32 + nt * 16 + l16;
                union { v8h v; _Float16 e[8]; } o;
                #pragma unroll
                for (int r = 0; r < 8; ++r) {
                    o.e[r] = (_Float16)acc[r];   // v_cvt_pk_f16_f32 pairs
                }
                o.v = __builtin_elementwise_max(o.v, hzero); // 4x v_pk_max_num_f16 ReLU
                *(v8h*)(H + p * HSTR + mt * 16 + 8 * half) = o.v;
            }
        }
    }

    // --- output layer: Linear(128,3), thread t owns row t; vector ds loads ---
    {
        const int p = blockIdx.x * BLK + t;
        float a0 = b_out[0], a1 = b_out[1], a2 = b_out[2];
        const _Float16* hr = H + t * HSTR;
        #pragma unroll
        for (int c = 0; c < 16; ++c) {
            v8h chunk = *(const v8h*)(hr + c * 8);
            #pragma unroll
            for (int e = 0; e < 8; ++e) {
                float hv = (float)chunk[e];
                int k = c * 8 + e;
                a0 = fmaf(hv, w_out[k * 3 + 0], a0);
                a1 = fmaf(hv, w_out[k * 3 + 1], a1);
                a2 = fmaf(hv, w_out[k * 3 + 2], a2);
            }
        }
        out[p * 3 + 0] = a0;
        out[p * 3 + 1] = a1;
        out[p * 3 + 2] = a2;
    }
}

extern "C" void kernel_launch(void* const* d_in, const int* in_sizes, int n_in,
                              void* d_out, int out_size, void* d_ws, size_t ws_size,
                              hipStream_t stream) {
    const float* x     = (const float*)d_in[0];
    const float* w_in  = (const float*)d_in[1];
    const float* b_in  = (const float*)d_in[2];
    const float* w_hid = (const float*)d_in[3];
    const float* b_hid = (const float*)d_in[4];
    const float* w_out = (const float*)d_in[5];
    const float* b_out = (const float*)d_in[6];
    float* out = (float*)d_out;

    _Float16* wt = (_Float16*)d_ws;  // 7*128*128*2 = 229,376 B of scratch

    const int n_w = N_HIDDEN * LATENT * LATENT;
    xpose_weights_f16<<<(n_w + 255) / 256, 256, 0, stream>>>(w_hid, wt);

    const size_t shmem = (size_t)LDS_HALFS * sizeof(_Float16); // 313,344 B
    static bool attr_set = [] {
        (void)hipFuncSetAttribute((const void*)mlp_wmma_kernel,
                                  hipFuncAttributeMaxDynamicSharedMemorySize,
                                  (int)((size_t)LDS_HALFS * sizeof(_Float16)));
        return true;
    }();
    (void)attr_set;

    const int n_pts = in_sizes[0] / 3; // N = 1048576 -> 4096 blocks of 256
    mlp_wmma_kernel<<<n_pts / BLK, BLK, shmem, stream>>>(
        x, w_in, b_in, b_hid, w_out, b_out, wt, out);
}
